// SingleHeadGATLayer_28827820491147
// MI455X (gfx1250) — compile-verified, matrix operand order checked by reading, not run on hardware
//
#include <hip/hip_runtime.h>
#include <stdint.h>

typedef __attribute__((ext_vector_type(16))) __bf16 v16bf;
typedef __attribute__((ext_vector_type(8)))  float  v8f;

#define IN_DIM   256
#define OUT_DIM  64
#define NEG_SLOPE 0.01f
#define NFRAG    (8 * 4 * 32 * 8)    // dwords in W fragment table

// ---- helpers -------------------------------------------------------------

// pack hi16(ra)|hi16(rb)<<16 in one v_perm_b32; inputs pre-rounded
__device__ __forceinline__ uint32_t bfmerge(uint32_t ra, uint32_t rb) {
    return __builtin_amdgcn_perm(rb, ra, 0x07060302u);  // [ra.b2,ra.b3,rb.b2,rb.b3]
}

// round-half-up fp32->bf16 pair pack: 3 VALU ops
__device__ __forceinline__ uint32_t pack_bf16(float a, float b) {
    return bfmerge(__float_as_uint(a) + 0x8000u, __float_as_uint(b) + 0x8000u);
}

union BF16x16 { uint32_t u[8]; uint4 q[2]; v16bf v; };

// monotone float->uint map so unsigned atomicMax == float max
__device__ __forceinline__ unsigned fflip(float f) {
    unsigned u = __float_as_uint(f);
    return (u & 0x80000000u) ? ~u : (u | 0x80000000u);
}
__device__ __forceinline__ float funflip(unsigned u) {
    return __uint_as_float((u & 0x80000000u) ? (u ^ 0x80000000u) : ~u);
}

// K-offset pattern for 16-bit A/B WMMA fragments (ISA 7.12.2)
__device__ __forceinline__ int kOff(int v, int half) {
    return (v < 4) ? (half * 8 + 2 * v) : (16 + half * 8 + 2 * (v - 4));
}

// ---- kernel 0: init out / segmax / denom ---------------------------------

__global__ void gat_init(float* __restrict__ out, unsigned* __restrict__ mmax,
                         float* __restrict__ denom, int N) {
    int i = blockIdx.x * blockDim.x + threadIdx.x;
    long long tot = (long long)N * OUT_DIM;
    if (i < tot) out[i] = 0.0f;
    if (i < N) { mmax[i] = 0u; denom[i] = 0.0f; }
}

// ---- kernel 0b: pack W into per-lane WMMA B fragments (once) -------------
// layout: [kt][nt][lane][8 dwords]

__global__ void gat_prepw(const float* __restrict__ W, uint32_t* __restrict__ wfrag) {
    int j = blockIdx.x * blockDim.x + threadIdx.x;
    if (j >= NFRAG) return;
    int v  = j & 7;
    int ln = (j >> 3) & 31;
    int nt = (j >> 8) & 3;
    int kt = j >> 10;
    int ko = kOff(v, ln >> 4);
    int k  = kt * 32 + ko;
    int n  = nt * 16 + (ln & 15);
    wfrag[j] = pack_bf16(W[k * OUT_DIM + n], W[(k + 1) * OUT_DIM + n]);
}

// ---- kernel 1: z = features @ W (bf16 WMMA) + fused el/er projection -----
// block = 128 threads = 4 waves; each wave: 16 rows x 64 cols.

__global__ void __launch_bounds__(128)
gat_gemm(const float* __restrict__ feat, const uint32_t* __restrict__ wfrag,
         const float* __restrict__ a, float* __restrict__ z,
         float* __restrict__ el, float* __restrict__ er, int N) {
    __shared__ uint4 sBf[NFRAG / 4];                   // 32 KB fragment table

    const int lane = threadIdx.x & 31;
    const int wave = threadIdx.x >> 5;
    const int l16  = lane & 15;
    const int half = lane >> 4;
    const int rowBase = blockIdx.x * 64;

    // flat coalesced copy of pre-packed W fragments into LDS
    const uint4* gw = (const uint4*)wfrag;
    for (int j = threadIdx.x; j < NFRAG / 4; j += 128) sBf[j] = gw[j];

    // A row this lane feeds; clamp (OOB rows computed but never stored)
    int row = rowBase + wave * 16 + l16;
    row = (row < N) ? row : (N - 1);
    // fold the half-dependent +8K offset into the base pointer
    const float* arow = feat + (size_t)row * IN_DIM + half * 8;
    __builtin_prefetch(arow, 0, 3);                    // global_prefetch_b8

    __syncthreads();

    v8f acc[4];
    #pragma unroll
    for (int nt = 0; nt < 4; ++nt)
        acc[nt] = (v8f){0.f,0.f,0.f,0.f,0.f,0.f,0.f,0.f};

    #pragma unroll
    for (int kt = 0; kt < IN_DIM / 32; ++kt) {
        const int kb = kt * 32;

        // A fragment: two contiguous 8-float runs -> 4x global_load_b128
        float4 f0 = *(const float4*)(arow + kb + 0);
        float4 f1 = *(const float4*)(arow + kb + 4);
        float4 f2 = *(const float4*)(arow + kb + 16);
        float4 f3 = *(const float4*)(arow + kb + 20);
        BF16x16 af;
        af.u[0] = pack_bf16(f0.x, f0.y);
        af.u[1] = pack_bf16(f0.z, f0.w);
        af.u[2] = pack_bf16(f1.x, f1.y);
        af.u[3] = pack_bf16(f1.z, f1.w);
        af.u[4] = pack_bf16(f2.x, f2.y);
        af.u[5] = pack_bf16(f2.z, f2.w);
        af.u[6] = pack_bf16(f3.x, f3.y);
        af.u[7] = pack_bf16(f3.z, f3.w);

        #pragma unroll
        for (int nt = 0; nt < 4; ++nt) {
            int fi = ((kt * 4 + nt) * 32 + lane) * 2;  // uint4 index
            BF16x16 bf;
            bf.q[0] = sBf[fi];                          // ds_load_b128
            bf.q[1] = sBf[fi + 1];                      // ds_load_b128
            acc[nt] = __builtin_amdgcn_wmma_f32_16x16x32_bf16(
                false, af.v, false, bf.v, (short)0, acc[nt], false, false);
        }
    }

    // attention vector slices for this lane's columns
    float aL[4], aR[4];
    #pragma unroll
    for (int nt = 0; nt < 4; ++nt) {
        aL[nt] = a[nt * 16 + l16];
        aR[nt] = a[OUT_DIM + nt * 16 + l16];
    }

    // C/D layout: VGPR r, lanes 0-15 -> M=r, lanes 16-31 -> M=r+8, N=lane%16
    float pel[8], per[8];
    #pragma unroll
    for (int r = 0; r < 8; ++r) { pel[r] = 0.f; per[r] = 0.f; }

    #pragma unroll
    for (int nt = 0; nt < 4; ++nt) {
        #pragma unroll
        for (int r = 0; r < 8; ++r) {
            int m = rowBase + wave * 16 + r + half * 8;
            float v = acc[nt][r];
            if (m < N) z[(size_t)m * OUT_DIM + nt * 16 + l16] = v;
            pel[r] += v * aL[nt];
            per[r] += v * aR[nt];
        }
    }

    // reduce el/er partials across the 16 lanes sharing each output row
    #pragma unroll
    for (int mask = 8; mask >= 1; mask >>= 1) {
        #pragma unroll
        for (int r = 0; r < 8; ++r) {
            pel[r] += __shfl_xor(pel[r], mask, 32);
            per[r] += __shfl_xor(per[r], mask, 32);
        }
    }
    if (l16 == 0) {
        #pragma unroll
        for (int r = 0; r < 8; ++r) {
            int m = rowBase + wave * 16 + r + half * 8;
            if (m < N) { el[m] = pel[r]; er[m] = per[r]; }
        }
    }
}

// ---- kernel 2: edge score + segment max ----------------------------------

__global__ void gat_edge_score(const float* __restrict__ el, const float* __restrict__ er,
                               const int* __restrict__ src, const int* __restrict__ dst,
                               float* __restrict__ ebuf, unsigned* __restrict__ mmax, int E) {
    int i = blockIdx.x * blockDim.x + threadIdx.x;
    if (i >= E) return;
    float e = el[src[i]] + er[dst[i]];
    e = (e > 0.f) ? e : NEG_SLOPE * e;            // leaky relu
    ebuf[i] = e;
    atomicMax(&mmax[dst[i]], fflip(e));           // global_atomic_max_u32
}

// ---- kernel 3: exp(e - m[dst]) + segment sum ------------------------------

__global__ void gat_edge_exp(const int* __restrict__ dst, const unsigned* __restrict__ mmax,
                             float* __restrict__ ebuf, float* __restrict__ denom, int E) {
    int i = blockIdx.x * blockDim.x + threadIdx.x;
    if (i >= E) return;
    int d = dst[i];
    float ex = __expf(ebuf[i] - funflip(mmax[d]));
    ebuf[i] = ex;
    atomicAdd(&denom[d], ex);                     // global_atomic_add_f32
}

// ---- kernel 4: h[dst] += alpha * z[src]  (64 threads per edge) -----------

__global__ void gat_aggregate(const int* __restrict__ src, const int* __restrict__ dst,
                              const float* __restrict__ ebuf, const float* __restrict__ denom,
                              const float* __restrict__ z, float* __restrict__ out, int E) {
    long long t = (long long)blockIdx.x * blockDim.x + threadIdx.x;
    int edge = (int)(t >> 6);
    int d    = (int)(t & 63);
    if (edge >= E) return;
    int dn = dst[edge], sn = src[edge];
    float den = denom[dn];
    den = (den == 0.f) ? 1.f : den;
    float alpha = ebuf[edge] / den;
    atomicAdd(&out[(size_t)dn * OUT_DIM + d], alpha * z[(size_t)sn * OUT_DIM + d]);
}

// ---- kernel 5: ELU --------------------------------------------------------

__global__ void gat_elu(float* __restrict__ out, long long tot) {
    long long i = (long long)blockIdx.x * blockDim.x + threadIdx.x;
    if (i >= tot) return;
    float h = out[i];
    out[i] = (h > 0.f) ? h : (__expf(h) - 1.f);
}

// ---- launcher -------------------------------------------------------------

extern "C" void kernel_launch(void* const* d_in, const int* in_sizes, int n_in,
                              void* d_out, int out_size, void* d_ws, size_t ws_size,
                              hipStream_t stream) {
    const float* feat = (const float*)d_in[0];
    const float* W    = (const float*)d_in[1];
    const float* a    = (const float*)d_in[2];
    const int*   src  = (const int*)d_in[3];
    const int*   dst  = (const int*)d_in[4];
    float* out = (float*)d_out;

    const int N = in_sizes[0] / IN_DIM;
    const int E = in_sizes[3];

    // workspace layout
    char* ws = (char*)d_ws;
    float*    z     = (float*)ws;                       // N*64 f32
    float*    el    = (float*)(ws + (size_t)N * OUT_DIM * 4);
    float*    er    = el + N;
    unsigned* mmax  = (unsigned*)(er + N);
    float*    denom = (float*)(mmax + N);
    float*    ebuf  = denom + N;                        // E f32
    uint32_t* wfrag = (uint32_t*)(ebuf + E);            // 32 KB fragment table

    const long long tot = (long long)N * OUT_DIM;

    gat_init<<<(unsigned)((tot + 255) / 256), 256, 0, stream>>>(out, mmax, denom, N);
    gat_prepw<<<NFRAG / 256, 256, 0, stream>>>(W, wfrag);
    gat_gemm<<<(N + 63) / 64, 128, 0, stream>>>(feat, wfrag, a, z, el, er, N);
    gat_edge_score<<<(E + 255) / 256, 256, 0, stream>>>(el, er, src, dst, ebuf, mmax, E);
    gat_edge_exp<<<(E + 255) / 256, 256, 0, stream>>>(dst, mmax, ebuf, denom, E);
    gat_aggregate<<<(unsigned)(((long long)E * 64 + 255) / 256), 256, 0, stream>>>(
        src, dst, ebuf, denom, z, out, E);
    gat_elu<<<(unsigned)((tot + 255) / 256), 256, 0, stream>>>(out, tot);
}
